// MVMamba_15874199126082
// MI455X (gfx1250) — matrix operand order repaired and checked by reading, not compiled, not run on hardware
//
#include <hip/hip_runtime.h>
#include <hip/hip_bf16.h>
#include <math.h>

#define L_SPATIAL 13824            // 24*24*24
#define NBRANCH   3
#define TOK3      (NBRANCH * L_SPATIAL)   // 41472 tokens (all 3 branches batched)
#define CDIM      64
#define DINNER_   128
#define DSTATE_   16
#define DTRANK_   4
#define XDBL_PAD  48               // 4 (dt) + 16 (B) + 16 (C) padded to 48 for WMMA N-tiles
#define NCHUNK    54               // 13824 / 256
#define LCHUNK    256
#define EPSV      1e-5f

typedef __bf16        v16bf __attribute__((ext_vector_type(16)));
typedef float         v8f   __attribute__((ext_vector_type(8)));
typedef unsigned int  v4u   __attribute__((ext_vector_type(4)));
typedef unsigned int  u32x4 __attribute__((ext_vector_type(4)));
typedef int           s32x4 __attribute__((ext_vector_type(4)));
typedef int           s32x8 __attribute__((ext_vector_type(8)));

union BF16Tile { v16bf v; v4u q[2]; };

__device__ __forceinline__ unsigned short f2bf(float f) {
  unsigned int u = __float_as_uint(f);
  u += 0x7FFFu + ((u >> 16) & 1u);       // round-to-nearest-even
  return (unsigned short)(u >> 16);
}
__device__ __forceinline__ float sigmoidf_(float x) { return 1.f / (1.f + __expf(-x)); }
__device__ __forceinline__ float siluf_(float x)    { return x * sigmoidf_(x); }
__device__ __forceinline__ float softplusf_(float x){ return (x > 20.f) ? x : log1pf(__expf(x)); }

// ---------------------------------------------------------------------------
// f32 -> bf16 weight conversion
// ---------------------------------------------------------------------------
__global__ void f32_to_bf16_kernel(const float* __restrict__ src,
                                   unsigned short* __restrict__ dst, int n) {
  int i = blockIdx.x * blockDim.x + threadIdx.x;
  if (i < n) dst[i] = f2bf(src[i]);
}

// ---------------------------------------------------------------------------
// 3 directional 3-tap convs (VALU; boundary-heavy) + fused per-channel
// instance-norm statistics. Output token-major vraw[b][l][c].
// grid = (54 l-blocks, 64 channels, 3 branches), block = 256 (one l each)
// ---------------------------------------------------------------------------
__global__ void conv_in_kernel(const float* __restrict__ x,
    const float* __restrict__ w1, const float* __restrict__ b1,
    const float* __restrict__ w2, const float* __restrict__ b2,
    const float* __restrict__ w3, const float* __restrict__ b3,
    float* __restrict__ vraw, float* __restrict__ ssum, float* __restrict__ ssq) {
  const int tid = threadIdx.x;
  const int c = blockIdx.y, br = blockIdx.z;
  const int l = blockIdx.x * 256 + tid;
  const float* w  = (br == 0) ? w1 : (br == 1) ? w2 : w3;   // all flatten to [c][ic][t]
  const float* bb = (br == 0) ? b1 : (br == 1) ? b2 : b3;
  int shift, coord;
  if (br == 0)      { shift = 576; coord = l / 576; }        // along D
  else if (br == 1) { shift = 24;  coord = (l / 24) % 24; }  // along H
  else              { shift = 1;   coord = l % 24; }         // along W
  const bool vm = coord > 0, vp = coord < 23;
  float acc = bb[c];
  const float* wc = w + c * 192;
  const float* xl = x + l;
  #pragma unroll 4
  for (int ic = 0; ic < 64; ++ic) {
    const float* xc = xl + ic * L_SPATIAL;
    float wa = wc[ic * 3 + 0], wbv = wc[ic * 3 + 1], wcv = wc[ic * 3 + 2];
    acc += wbv * xc[0];
    if (vm) acc += wa  * xc[-shift];
    if (vp) acc += wcv * xc[shift];
  }
  vraw[((size_t)br * L_SPATIAL + l) * 64 + c] = acc;

  __shared__ float s1[256]; __shared__ float s2[256];
  s1[tid] = acc; s2[tid] = acc * acc; __syncthreads();
  for (int s = 128; s > 0; s >>= 1) {
    if (tid < s) { s1[tid] += s1[tid + s]; s2[tid] += s2[tid + s]; }
    __syncthreads();
  }
  if (tid == 0) {
    atomicAdd(&ssum[br * 64 + c], s1[0]);
    atomicAdd(&ssq [br * 64 + c], s2[0]);
  }
}

__global__ void finalize_stats_kernel(const float* __restrict__ ssum,
    const float* __restrict__ ssq, float* __restrict__ mean,
    float* __restrict__ rstd, int n, float inv_count) {
  int i = blockIdx.x * blockDim.x + threadIdx.x;
  if (i < n) {
    float m = ssum[i] * inv_count;
    float v = ssq[i] * inv_count - m * m;
    mean[i] = m; rstd[i] = rsqrtf(v + EPSV);
  }
}

// ---------------------------------------------------------------------------
// Fused instance-norm apply + ReLU + LayerNorm(64) -> bf16 tokens for WMMA.
// block = 256 = 4 tokens x 64 channels (wave32-aware shuffle reduction)
// ---------------------------------------------------------------------------
__global__ void in_relu_ln_kernel(const float* __restrict__ vraw,
    const float* __restrict__ mean, const float* __restrict__ rstd,
    const float* __restrict__ g1, const float* __restrict__ be1,
    const float* __restrict__ g2, const float* __restrict__ be2,
    const float* __restrict__ g3, const float* __restrict__ be3,
    const float* __restrict__ ln_w, const float* __restrict__ ln_b,
    unsigned short* __restrict__ xn_bf) {
  const int tid = threadIdx.x;
  const int tok = blockIdx.x * 4 + (tid >> 6);
  const int c = tid & 63;
  const int b = tok / L_SPATIAL;
  const float* g  = (b == 0) ? g1  : (b == 1) ? g2  : g3;
  const float* be = (b == 0) ? be1 : (b == 1) ? be2 : be3;
  float v = vraw[(size_t)tok * 64 + c];
  v = (v - mean[b * 64 + c]) * rstd[b * 64 + c] * g[c] + be[c];
  v = fmaxf(v, 0.f);
  float s = v, q = v * v;
  for (int m = 16; m >= 1; m >>= 1) { s += __shfl_xor(s, m); q += __shfl_xor(q, m); }
  __shared__ float shs[4][2]; __shared__ float shq[4][2];
  const int t = tid >> 6, half = (tid >> 5) & 1;
  if ((tid & 31) == 0) { shs[t][half] = s; shq[t][half] = q; }
  __syncthreads();
  float tot = shs[t][0] + shs[t][1], totq = shq[t][0] + shq[t][1];
  float mu  = tot * (1.f / 64.f);
  float var = totq * (1.f / 64.f) - mu * mu;
  float xn = (v - mu) * rsqrtf(var + EPSV) * ln_w[c] + ln_b[c];
  xn_bf[(size_t)tok * 64 + c] = f2bf(xn);
}

// ---------------------------------------------------------------------------
// bf16 WMMA GEMM with TDM-staged weights:
//   Out[M,N](f32) = A[M,K](bf16 row-major) @ W[N,K](bf16)^T
// Block = 8 waves -> a 128x16 output patch (8 M-tiles, one N-tile).
// Wave 0 DMAs the 16xK bf16 weight strip into LDS with TENSOR_LOAD_TO_LDS
// (D# per CDNA5 ISA), waits on TENSORcnt, then
// a workgroup barrier publishes it; all waves stream B from LDS (ds_load)
// while A comes in via coalesced global_load_b128. K templated (64/128/192)
// so the WMMA K-loop fully unrolls.
// ---------------------------------------------------------------------------
template <int K>
__global__ void gemm_bf16_wmma_kernel(const unsigned short* __restrict__ A,
    const unsigned short* __restrict__ W, const float* __restrict__ bias,
    float* __restrict__ Out, int N) {
  __shared__ unsigned short shW[16 * 192];   // max K = 192 -> 6 KB
  const int tid = threadIdx.x;
  const int wave = tid >> 5, lane = tid & 31;
  const int lo = lane & 15, half = lane >> 4;
  const int tN = blockIdx.y;

  if (wave == 0) {
#if defined(__HIP_DEVICE_COMPILE__)
    const unsigned long long gaddr =
        (unsigned long long)(const void*)(W + (size_t)tN * 16 * K);
    const unsigned int ldsAddr = (unsigned int)(size_t)(void*)shW;
    // D# group 0: count=1 | lds_addr | global_addr[56:0] | type=2
    u32x4 g0;
    g0[0] = 1u;
    g0[1] = ldsAddr;
    g0[2] = (unsigned int)(gaddr & 0xFFFFFFFFull);
    g0[3] = ((unsigned int)(gaddr >> 32) & 0x01FFFFFFu) | 0x80000000u;
    // D# group 1: data_size=1 (2B), tensor 16 rows x K, tile 16 x K,
    // tensor_dim0_stride = K elements, no multicast / padding / iterate.
    s32x8 g1;
    g1[0] = (int)(1u << 16);                 // workgroup_mask=0 | data_size=1
    g1[1] = (int)((unsigned)K << 16);        // tensor_dim0[15:0] = K
    g1[2] = (int)(16u << 16);                // tensor_dim0[31:16]=0 | tensor_dim1=16
    g1[3] = (int)((unsigned)K << 16);        // tensor_dim1[31:16]=0 | tile_dim0=K
    g1[4] = 16;                              // tile_dim1=16 | tile_dim2=0
    g1[5] = K;                               // tensor_dim0_stride[31:0]
    g1[6] = 0;                               // stride hi | tensor_dim1_stride lo
    g1[7] = 0;
    s32x4 z4 = {0, 0, 0, 0};
#if __clang_major__ >= 23
    s32x8 z8 = {0, 0, 0, 0, 0, 0, 0, 0};
    __builtin_amdgcn_tensor_load_to_lds(g0, g1, z4, z4, z8, 0);
#else
    __builtin_amdgcn_tensor_load_to_lds(g0, g1, z4, z4, 0);
#endif
    __builtin_amdgcn_s_wait_tensorcnt(0);
#endif
  }
  __syncthreads();

  const int tM = blockIdx.x * 8 + wave;
  v8f acc = {0.f, 0.f, 0.f, 0.f, 0.f, 0.f, 0.f, 0.f};
  const unsigned short* aRow = A + (size_t)(tM * 16 + lo) * K + half * 16;
  const unsigned short* wRow = shW + lo * K + half * 16;
  #pragma unroll
  for (int k0 = 0; k0 < K; k0 += 32) {
    BF16Tile a, b;
    a.q[0] = *(const v4u*)(aRow + k0);
    a.q[1] = *(const v4u*)(aRow + k0 + 8);
    b.q[0] = *(const v4u*)(wRow + k0);       // ds_load_b128 from staged weights
    b.q[1] = *(const v4u*)(wRow + k0 + 8);
    acc = __builtin_amdgcn_wmma_f32_16x16x32_bf16(
        false, a.v, false, b.v, (short)0, acc, false, false);
  }
  const int gn = tN * 16 + lo;
  const float bv = bias ? bias[gn] : 0.f;
  float* outp = Out + (size_t)(tM * 16 + half * 8) * N + gn;
  #pragma unroll
  for (int r = 0; r < 8; ++r) outp[(size_t)r * N] = acc[r] + bv;
}

// xi = silu(xz[:, :128] * conv_w + conv_b) ; keep f32 (for scan) and bf16 (x_proj A)
__global__ void silu_conv_kernel(const float* __restrict__ xz,
    const float* __restrict__ conv_w, const float* __restrict__ conv_b,
    float* __restrict__ xi_f, unsigned short* __restrict__ xi_bf) {
  int idx = blockIdx.x * 256 + threadIdx.x;
  int tok = idx >> 7, i = idx & 127;
  float v = xz[(size_t)tok * 256 + i];
  v = siluf_(v * conv_w[i] + conv_b[i]);
  xi_f[idx] = v;
  xi_bf[idx] = f2bf(v);
}

// ---------------------------------------------------------------------------
// Chunked selective scan. 2048 independent (i,n) recurrences per branch.
// Pass1: per-chunk (prod dA, local h).  Pass3: full scan from chunk prefix h0,
// fused with y = <h, C> + xi*D and SiLU(z) gating (shfl reduction over the
// 16 state-lanes of each wave32 half).
// grid = (NCHUNK, 8 channel-groups, 3 branches), block = 256 = 16 ch x 16 st
// ---------------------------------------------------------------------------
template <bool PASS3>
__global__ void scan_kernel(const float* __restrict__ xdbl,
    const float* __restrict__ xi_f, const float* __restrict__ xz,
    const float* __restrict__ dt_w, const float* __restrict__ dt_b,
    const float* __restrict__ A_log, const float* __restrict__ D_param,
    float* __restrict__ chunkA, float* __restrict__ chunkH,
    const float* __restrict__ h0, unsigned short* __restrict__ y_bf) {
  const int tid = threadIdx.x;
  const int chunk = blockIdx.x, ig = blockIdx.y, b = blockIdx.z;
  const int i = ig * 16 + (tid >> 4);
  const int n = tid & 15;
  const float Aneg = -__expf(A_log[i * 16 + n]);
  const float w0 = dt_w[i * 4 + 0], w1 = dt_w[i * 4 + 1];
  const float w2 = dt_w[i * 4 + 2], w3 = dt_w[i * 4 + 3];
  const float db = dt_b[i];
  const float Dp = D_param[i];
  const size_t cidx = ((size_t)b * NCHUNK + chunk) * 2048 + (size_t)i * 16 + n;
  float aP = 1.f, h = 0.f;
  if (PASS3) h = h0[cidx];
  const size_t tok0 = (size_t)b * L_SPATIAL + (size_t)chunk * LCHUNK;
  for (int l = 0; l < LCHUNK; ++l) {
    const size_t tok = tok0 + l;
    const float* row = xdbl + tok * XDBL_PAD;
    __builtin_prefetch(row + XDBL_PAD, 0, 0);   // global_prefetch_b8 next token
    float dt = softplusf_(row[0] * w0 + row[1] * w1 + row[2] * w2 + row[3] * w3 + db);
    float Bm  = row[DTRANK_ + n];
    float xiv = xi_f[tok * DINNER_ + i];
    float dA  = __expf(dt * Aneg);
    float dBx = dt * Bm * xiv;
    h = dA * h + dBx;
    if (!PASS3) {
      aP *= dA;
    } else {
      float Cm = row[DTRANK_ + DSTATE_ + n];
      float py = h * Cm;
      py += __shfl_xor(py, 1);
      py += __shfl_xor(py, 2);
      py += __shfl_xor(py, 4);
      py += __shfl_xor(py, 8);
      if (n == 0) {
        float yv = py + xiv * Dp;
        float z = xz[tok * 256 + 128 + i];
        y_bf[tok * DINNER_ + i] = f2bf(yv * siluf_(z));
      }
    }
  }
  if (!PASS3) { chunkA[cidx] = aP; chunkH[cidx] = h; }
}

// Pass2: sequential combine over the 54 chunk summaries per (b,i,n).
__global__ void scan_combine_kernel(const float* __restrict__ chunkA,
    const float* __restrict__ chunkH, float* __restrict__ h0) {
  int idx = blockIdx.x * 256 + threadIdx.x;   // 0 .. 3*2048-1
  int b = idx >> 11;
  int j = idx & 2047;
  float hp = 0.f;
  for (int c = 0; c < NCHUNK; ++c) {
    size_t o = ((size_t)b * NCHUNK + c) * 2048 + j;
    h0[o] = hp;
    hp = chunkA[o] * hp + chunkH[o];
  }
}

// Concat the 3 branch out_proj results into token-major bf16 (L x 192)
__global__ void build_cat_kernel(const float* __restrict__ m_f,
                                 unsigned short* __restrict__ cat_bf) {
  int idx = blockIdx.x * 256 + threadIdx.x;   // 3*L*64
  int b = idx / (L_SPATIAL * 64);
  int rem = idx - b * (L_SPATIAL * 64);
  int l = rem >> 6, c = rem & 63;
  cat_bf[(size_t)l * 192 + b * 64 + c] = f2bf(m_f[idx]);
}

// Per-channel stats for final instance norm (one block per channel)
__global__ void colstats_kernel(const float* __restrict__ f_raw,
    float* __restrict__ fsum, float* __restrict__ fsq) {
  int c = blockIdx.x, tid = threadIdx.x;
  float s = 0.f, q = 0.f;
  for (int l = tid; l < L_SPATIAL; l += 256) {
    float v = f_raw[(size_t)l * 64 + c];
    s += v; q += v * v;
  }
  __shared__ float s1[256]; __shared__ float s2[256];
  s1[tid] = s; s2[tid] = q; __syncthreads();
  for (int st = 128; st > 0; st >>= 1) {
    if (tid < st) { s1[tid] += s1[tid + st]; s2[tid] += s2[tid + st]; }
    __syncthreads();
  }
  if (tid == 0) { fsum[c] = s1[0]; fsq[c] = s2[0]; }
}

// Final IN + ReLU, transposed back to channel-major output (1,64,24,24,24)
__global__ void final_apply_kernel(const float* __restrict__ f_raw,
    const float* __restrict__ fmean, const float* __restrict__ frstd,
    const float* __restrict__ gf, const float* __restrict__ bef,
    float* __restrict__ out) {
  int idx = blockIdx.x * 256 + threadIdx.x;
  int c = idx / L_SPATIAL, l = idx - c * L_SPATIAL;
  float v = f_raw[(size_t)l * 64 + c];
  v = (v - fmean[c]) * frstd[c] * gf[c] + bef[c];
  out[idx] = fmaxf(v, 0.f);
}

// ---------------------------------------------------------------------------
extern "C" void kernel_launch(void* const* d_in, const int* in_sizes, int n_in,
                              void* d_out, int out_size, void* d_ws, size_t ws_size,
                              hipStream_t stream) {
  (void)in_sizes; (void)n_in; (void)out_size; (void)ws_size;
  const float* x        = (const float*)d_in[0];
  const float* w1       = (const float*)d_in[1];
  const float* b1       = (const float*)d_in[2];
  const float* g1       = (const float*)d_in[3];
  const float* be1      = (const float*)d_in[4];
  const float* w2       = (const float*)d_in[5];
  const float* b2       = (const float*)d_in[6];
  const float* g2       = (const float*)d_in[7];
  const float* be2      = (const float*)d_in[8];
  const float* w3       = (const float*)d_in[9];
  const float* b3       = (const float*)d_in[10];
  const float* g3       = (const float*)d_in[11];
  const float* be3      = (const float*)d_in[12];
  const float* ln_w     = (const float*)d_in[13];
  const float* ln_b     = (const float*)d_in[14];
  const float* in_proj_w= (const float*)d_in[15];
  const float* conv_w   = (const float*)d_in[16];
  const float* conv_b   = (const float*)d_in[17];
  const float* x_proj_w = (const float*)d_in[18];
  const float* dt_w     = (const float*)d_in[19];
  const float* dt_b     = (const float*)d_in[20];
  const float* A_log    = (const float*)d_in[21];
  const float* D_param  = (const float*)d_in[22];
  const float* out_proj_w = (const float*)d_in[23];
  const float* wf       = (const float*)d_in[24];
  const float* bf       = (const float*)d_in[25];
  const float* gf       = (const float*)d_in[26];
  const float* bef      = (const float*)d_in[27];

  // --- workspace carve-out (256B aligned), with buffer overlays ---
  char* p = (char*)d_ws;
  auto alloc = [&](size_t bytes) -> char* {
    char* r = p; p += (bytes + 255) & ~(size_t)255; return r;
  };
  float* vraw           = (float*)alloc((size_t)TOK3 * 64 * 4);     // reused as m_f
  unsigned short* xn_bf = (unsigned short*)alloc((size_t)TOK3 * 64 * 2); // reused as cat_bf
  float* xz             = (float*)alloc((size_t)TOK3 * 256 * 4);
  float* xi_f           = (float*)alloc((size_t)TOK3 * 128 * 4);
  unsigned short* xi_bf = (unsigned short*)alloc((size_t)TOK3 * 128 * 2);
  float* xdbl           = (float*)alloc((size_t)TOK3 * XDBL_PAD * 4);  // reused as f_raw
  float* chunkA         = (float*)alloc((size_t)NBRANCH * NCHUNK * 2048 * 4);
  float* chunkH         = (float*)alloc((size_t)NBRANCH * NCHUNK * 2048 * 4);
  float* h0s            = (float*)alloc((size_t)NBRANCH * NCHUNK * 2048 * 4);
  unsigned short* y_bf  = (unsigned short*)alloc((size_t)TOK3 * 128 * 2);
  unsigned short* ipw   = (unsigned short*)alloc(256 * 64 * 2);
  unsigned short* xpw   = (unsigned short*)alloc(48 * 128 * 2);
  unsigned short* opw   = (unsigned short*)alloc(64 * 128 * 2);
  unsigned short* wfb   = (unsigned short*)alloc(64 * 192 * 2);
  float* stats          = (float*)alloc(1024 * 4);
  float* m_f   = vraw;                       // overlay: vraw dead after in_ln
  unsigned short* cat_bf = xn_bf;            // overlay: xn dead after in_proj GEMM
  float* f_raw = xdbl;                       // overlay: xdbl dead after scan pass3
  float* ssum = stats,       *ssq = stats + 192;
  float* smean = stats + 384, *srstd = stats + 576;
  float* fsum = stats + 768,  *fsq = stats + 832;
  float* fmean = stats + 896, *frstd = stats + 960;

  hipMemsetAsync(stats, 0, 1024 * 4, stream);
  hipMemsetAsync(xpw + 36 * 128, 0, 12 * 128 * 2, stream);  // zero-pad x_proj rows 36..47

  // weights -> bf16
  f32_to_bf16_kernel<<<64, 256, 0, stream>>>(in_proj_w, ipw, 256 * 64);
  f32_to_bf16_kernel<<<18, 256, 0, stream>>>(x_proj_w, xpw, 36 * 128);
  f32_to_bf16_kernel<<<32, 256, 0, stream>>>(out_proj_w, opw, 64 * 128);
  f32_to_bf16_kernel<<<48, 256, 0, stream>>>(wf, wfb, 64 * 192);

  // 3 directional convs + IN stats
  conv_in_kernel<<<dim3(NCHUNK, 64, 3), 256, 0, stream>>>(
      x, w1, b1, w2, b2, w3, b3, vraw, ssum, ssq);
  finalize_stats_kernel<<<1, 256, 0, stream>>>(ssum, ssq, smean, srstd, 192,
                                               1.f / (float)L_SPATIAL);

  // IN apply + ReLU + LN -> bf16 tokens
  in_relu_ln_kernel<<<TOK3 / 4, 256, 0, stream>>>(
      vraw, smean, srstd, g1, be1, g2, be2, g3, be3, ln_w, ln_b, xn_bf);

  // in_proj: (41472 x 64) @ (64 x 256) -> xz
  gemm_bf16_wmma_kernel<64><<<dim3(TOK3 / 128, 256 / 16), 256, 0, stream>>>(
      xn_bf, ipw, nullptr, xz, 256);

  // xi = silu(x * conv_w + conv_b)
  silu_conv_kernel<<<(TOK3 * 128) / 256, 256, 0, stream>>>(
      xz, conv_w, conv_b, xi_f, xi_bf);

  // x_proj: (41472 x 128) @ (128 x 48pad) -> xdbl
  gemm_bf16_wmma_kernel<128><<<dim3(TOK3 / 128, 48 / 16), 256, 0, stream>>>(
      xi_bf, xpw, nullptr, xdbl, XDBL_PAD);

  // chunked selective scan (3 passes)
  scan_kernel<false><<<dim3(NCHUNK, 8, 3), 256, 0, stream>>>(
      xdbl, xi_f, nullptr, dt_w, dt_b, A_log, D_param, chunkA, chunkH, nullptr, nullptr);
  scan_combine_kernel<<<(NBRANCH * 2048) / 256, 256, 0, stream>>>(chunkA, chunkH, h0s);
  scan_kernel<true><<<dim3(NCHUNK, 8, 3), 256, 0, stream>>>(
      xdbl, xi_f, xz, dt_w, dt_b, A_log, D_param, nullptr, nullptr, h0s, y_bf);

  // out_proj: (41472 x 128) @ (128 x 64) -> m_f
  gemm_bf16_wmma_kernel<128><<<dim3(TOK3 / 128, 64 / 16), 256, 0, stream>>>(
      y_bf, opw, nullptr, m_f, 64);

  // concat 3 branches into (L x 192) bf16
  build_cat_kernel<<<(TOK3 * 64) / 256, 256, 0, stream>>>(m_f, cat_bf);

  // final 1x1x1 conv: (13824 x 192) @ (192 x 64) + bias -> f_raw
  gemm_bf16_wmma_kernel<192><<<dim3(L_SPATIAL / 128, 64 / 16), 256, 0, stream>>>(
      cat_bf, wfb, bf, f_raw, 64);

  // final instance norm + ReLU -> channel-major output
  colstats_kernel<<<64, 256, 0, stream>>>(f_raw, fsum, fsq);
  finalize_stats_kernel<<<1, 64, 0, stream>>>(fsum, fsq, fmean, frstd, 64,
                                              1.f / (float)L_SPATIAL);
  final_apply_kernel<<<(64 * L_SPATIAL) / 256, 256, 0, stream>>>(
      f_raw, fmean, frstd, gf, bef, (float*)d_out);
}
